// FullyConnectedModel_21723944583762
// MI455X (gfx1250) — matrix operand order repaired and checked by reading, not compiled, not run on hardware
//
#include <hip/hip_runtime.h>
#include <hip/hip_bf16.h>
#include <math.h>

// ---------------------------------------------------------------------------
// Model dims (compile-time constants from the reference)
// ---------------------------------------------------------------------------
#define BATCH       64
#define SENSORS     256
#define ROWS        (BATCH * SENSORS)      // 16384
#define WID         1024                   // hidden width
#define DEPTH       8
#define UFLAT       1280                   // SENSORS * (C + D)
#define INSZ        1282                   // UFLAT + 2 coords
#define NCH         3

typedef __attribute__((ext_vector_type(16))) __bf16 v16bf;
typedef __attribute__((ext_vector_type(8)))  float  v8f;

union Frag16 { v16bf v; uint4 q[2]; };

// Hardware tanh if the toolchain exposes it (gfx1250 V_TANH_F32), else OCML.
__device__ __forceinline__ float fast_tanh(float x)
{
#if __has_builtin(__builtin_amdgcn_tanhf)
  return __builtin_amdgcn_tanhf(x);
#elif __has_builtin(__builtin_amdgcn_tanh_f32)
  return __builtin_amdgcn_tanh_f32(x);
#else
  return tanhf(x);
#endif
}

// ---------------------------------------------------------------------------
// 0) Convert hidden weights fp32 -> bf16 (8M elems, trivial)
// ---------------------------------------------------------------------------
__global__ __launch_bounds__(256)
void fcnet_convw(const float* __restrict__ W, __bf16* __restrict__ Wb, int n)
{
  int i = (blockIdx.x * 256 + threadIdx.x) * 4;
  if (i < n) {
    float4 f = *(const float4*)(W + i);
    Wb[i + 0] = (__bf16)f.x;
    Wb[i + 1] = (__bf16)f.y;
    Wb[i + 2] = (__bf16)f.z;
    Wb[i + 3] = (__bf16)f.w;
  }
}

// ---------------------------------------------------------------------------
// 1) First layer, u-part only: t[b][w] = b_first[w] + sum_k u_flat[b][k]*Wf[w][k]
//    (all 256 query points in a batch share u_flat -> 43 GFLOP collapses to 0.17)
// ---------------------------------------------------------------------------
__global__ __launch_bounds__(256)
void fcnet_first_proj(const float* __restrict__ u, const float* __restrict__ Wf,
                      const float* __restrict__ bf, float* __restrict__ t)
{
  __shared__ float su[UFLAT];
  const int b = blockIdx.x;
  for (int k = threadIdx.x; k < UFLAT; k += 256)
    su[k] = u[b * UFLAT + k];
  __syncthreads();

  const int w = blockIdx.y * 256 + threadIdx.x;
  const float* wr = Wf + (size_t)w * INSZ;
  float acc = bf[w];
#pragma unroll 8
  for (int k = 0; k < UFLAT; ++k)
    acc += su[k] * wr[k];
  t[b * WID + w] = acc;
}

// ---------------------------------------------------------------------------
// 2) Expand: H[row][w] = t[b][w] + x0*Wf[w][1280] + x1*Wf[w][1281]
// ---------------------------------------------------------------------------
__global__ __launch_bounds__(256)
void fcnet_expand(const float* __restrict__ x, const float* __restrict__ Wf,
                  const float* __restrict__ t, float* __restrict__ H)
{
  const int i   = blockIdx.x * 256 + threadIdx.x;   // < ROWS*WID
  const int w   = i & (WID - 1);
  const int row = i >> 10;
  const int b   = row >> 8;
  const float x0 = x[row * 2 + 0];
  const float x1 = x[row * 2 + 1];
  H[i] = t[b * WID + w] + x0 * Wf[(size_t)w * INSZ + UFLAT]
                        + x1 * Wf[(size_t)w * INSZ + UFLAT + 1];
}

// ---------------------------------------------------------------------------
// 3) Hidden layer (WMMA, in place): H = tanh(H @ W^T + b) + H
//    One block per 64 rows; block computes all 1024 output cols.
//    A tile staged to LDS as bf16; residual read in fp32 before overwrite.
// ---------------------------------------------------------------------------
#define MT    64
#define ASTR  (WID + 8)   // bf16 units; row stride 2064 B (16B aligned, bank-staggered)
#define HID_LDS_BYTES (MT * ASTR * 2)

__global__ __launch_bounds__(256)
void fcnet_hidden_wmma(float* __restrict__ H, const __bf16* __restrict__ Wgt,
                       const float* __restrict__ bias)
{
  extern __shared__ __bf16 As[];            // [MT][ASTR]
  const int m0  = blockIdx.x * MT;
  const int tid = threadIdx.x;

  // Stage A tile (fp32 -> bf16) into LDS.
  for (int idx = tid * 4; idx < MT * WID; idx += 256 * 4) {
    const int row = idx >> 10;
    const int col = idx & (WID - 1);
    const float4 f = *(const float4*)(H + (size_t)(m0 + row) * WID + col);
    __bf16* d = As + row * ASTR + col;
    d[0] = (__bf16)f.x; d[1] = (__bf16)f.y; d[2] = (__bf16)f.z; d[3] = (__bf16)f.w;
  }
  __syncthreads();

  const int wave = tid >> 5;
  const int lane = tid & 31;
  const int l16  = lane & 15;
  const int hi   = lane >> 4;               // half-wave select

  const v8f vzero = {0.f, 0.f, 0.f, 0.f, 0.f, 0.f, 0.f, 0.f};

  for (int pass = 0; pass < 2; ++pass) {
    const int nbase = wave * 128 + pass * 64;   // this wave's 64-col output tile
    v8f acc[4][4];
#pragma unroll
    for (int mf = 0; mf < 4; ++mf)
#pragma unroll
      for (int nf = 0; nf < 4; ++nf)
        acc[mf][nf] = vzero;

    for (int kk = 0; kk < WID; kk += 32) {
      Frag16 fa[4], fb[4];
      // A frag (16x32 bf16): elem e -> K = kk + hi*8 + e  (+16 for e>=8), row M = l16
      const int ka = kk + hi * 8;
#pragma unroll
      for (int mf = 0; mf < 4; ++mf) {
        const __bf16* p = As + (mf * 16 + l16) * ASTR + ka;
        fa[mf].q[0] = *(const uint4*)p;          // K = ka .. ka+7
        fa[mf].q[1] = *(const uint4*)(p + 16);   // K = ka+16 .. ka+23
      }
      // B frag (32x16 bf16): elem e -> K = kk + hi*8 + e  (+16 for e>=8), col N = l16.
      // B[k][n] = W[n][k] (einsum vw), so this is two 16-byte chunks of W's row.
      const int kb = kk + hi * 8;
#pragma unroll
      for (int nf = 0; nf < 4; ++nf) {
        const __bf16* p = Wgt + (size_t)(nbase + nf * 16 + l16) * WID + kb;
        fb[nf].q[0] = *(const uint4*)p;          // K = kb .. kb+7
        fb[nf].q[1] = *(const uint4*)(p + 16);   // K = kb+16 .. kb+23
      }
#pragma unroll
      for (int mf = 0; mf < 4; ++mf)
#pragma unroll
        for (int nf = 0; nf < 4; ++nf)
          acc[mf][nf] = __builtin_amdgcn_wmma_f32_16x16x32_bf16(
              false, fa[mf].v, false, fb[nf].v, (short)0, acc[mf][nf],
              false, false);
    }

    // Epilogue: H = tanh(acc + bias) + H_old (fp32 residual; rows are
    // block-exclusive and cols are wave/pass-exclusive -> in-place safe).
#pragma unroll
    for (int nf = 0; nf < 4; ++nf) {
      const int   n  = nbase + nf * 16 + l16;
      const float bn = bias[n];
#pragma unroll
      for (int mf = 0; mf < 4; ++mf) {
        const int mr = m0 + mf * 16 + hi * 8;   // C/D: vgpr r -> M = r (+8 for hi half)
#pragma unroll
        for (int r = 0; r < 8; ++r) {
          float* hp = H + (size_t)(mr + r) * WID + n;
          const float hold = *hp;
          *hp = fast_tanh(acc[mf][nf][r] + bn) + hold;
        }
      }
    }
  }
}

// ---------------------------------------------------------------------------
// 4) Head: v[row][c] = b_last[c] + sum_k H[row][k] * W_last[c][k]
// ---------------------------------------------------------------------------
__global__ __launch_bounds__(256)
void fcnet_last(const float* __restrict__ H, const float* __restrict__ Wl,
                const float* __restrict__ bl, float* __restrict__ out)
{
  const int i   = blockIdx.x * 256 + threadIdx.x;   // < ROWS*NCH
  const int row = i / NCH;
  const int c   = i - row * NCH;
  const float* h = H  + (size_t)row * WID;
  const float* w = Wl + (size_t)c   * WID;
  float acc = bl[c];
  for (int k = 0; k < WID; k += 4)
    acc += h[k] * w[k] + h[k + 1] * w[k + 1] + h[k + 2] * w[k + 2] + h[k + 3] * w[k + 3];
  out[i] = acc;
}

// ---------------------------------------------------------------------------
// Launch
// ---------------------------------------------------------------------------
extern "C" void kernel_launch(void* const* d_in, const int* in_sizes, int n_in,
                              void* d_out, int out_size, void* d_ws, size_t ws_size,
                              hipStream_t stream)
{
  (void)in_sizes; (void)n_in; (void)out_size; (void)ws_size;

  const float* u   = (const float*)d_in[0];   // [64,256,5]
  const float* x   = (const float*)d_in[1];   // [64,256,2]
  const float* Wf  = (const float*)d_in[2];   // [1024,1282]
  const float* bf  = (const float*)d_in[3];   // [1024]
  const float* Wh  = (const float*)d_in[4];   // [8,1024,1024]
  const float* bh  = (const float*)d_in[5];   // [8,1024]
  const float* Wl  = (const float*)d_in[6];   // [3,1024]
  const float* bl  = (const float*)d_in[7];   // [3]
  float*       out = (float*)d_out;           // [64,256,3]

  // Workspace layout: H (64MB fp32) | Wb (16MB bf16) | t (256KB fp32)
  char*   ws = (char*)d_ws;
  float*  H  = (float*)ws;
  __bf16* Wb = (__bf16*)(ws + (size_t)ROWS * WID * sizeof(float));
  float*  t  = (float*) (ws + (size_t)ROWS * WID * sizeof(float)
                            + (size_t)DEPTH * WID * WID * sizeof(__bf16));

  // Allow >64KB dynamic LDS for the WMMA kernel (CDNA5: up to 320KB/WG).
  (void)hipFuncSetAttribute((const void*)fcnet_hidden_wmma,
                            hipFuncAttributeMaxDynamicSharedMemorySize,
                            HID_LDS_BYTES);

  const int nw = DEPTH * WID * WID;
  fcnet_convw<<<(nw / 4 + 255) / 256, 256, 0, stream>>>(Wh, Wb, nw);
  fcnet_first_proj<<<dim3(BATCH, WID / 256), 256, 0, stream>>>(u, Wf, bf, t);
  fcnet_expand<<<(ROWS * WID) / 256, 256, 0, stream>>>(x, Wf, t, H);

  for (int l = 0; l < DEPTH; ++l)
    fcnet_hidden_wmma<<<ROWS / MT, 256, HID_LDS_BYTES, stream>>>(
        H, Wb + (size_t)l * WID * WID, bh + (size_t)l * WID);

  fcnet_last<<<(ROWS * NCH) / 256, 256, 0, stream>>>(H, Wl, bl, out);
}
